// PatchEmbedding_8452495639010
// MI455X (gfx1250) — matrix operand order repaired and checked by reading, not compiled, not run on hardware
//
#include <hip/hip_runtime.h>
#include <hip/hip_bf16.h>

// ---------------------------------------------------------------------------
// PatchEmbedding on MI455X (gfx1250, wave32, WMMA).
// Combined conv: out[b,l,h] = sum_{k<1024} A[b,l,k] * Weff[h,k] + beff[h] + pos[l,h]
//   A[b,l,j*64+d] = x[b,l+j,d]   (sliding windows -> im2col-free GEMM)
//   Weff = (W2 + pad(W1) + pad(W0)) / 3,  beff = (b0+b1+b2)/3
// Precision: split-bf16 (hi+lo) with 3-product compensation on
// v_wmma_f32_16x16x32_bf16 -> ~fp32 accuracy at bf16 matrix-pipe speed.
// B k-tiles staged to LDS via GLOBAL_LOAD_ASYNC_TO_LDS (ASYNCcnt) when the
// toolchain exposes the builtin; register-prefetch + ds_store fallback else.
// ---------------------------------------------------------------------------

typedef __attribute__((ext_vector_type(16))) __bf16 v16bf;
typedef __attribute__((ext_vector_type(8)))  __bf16 v8bf;
typedef __attribute__((ext_vector_type(8)))  float  v8f;
typedef __attribute__((ext_vector_type(4)))  int    v4i;

#define S_LEN   1000
#define B_SZ    32
#define D_IN    64
#define H_OUT   512
#define KDIM    1024          // 16 taps * 64 channels
#define MINLEN  985           // 1000 - 16 + 1

#define M_BLOCK 128           // rows of 'l' per workgroup (8 waves x 16)
#define N_BLOCK 64            // 'h' columns per workgroup (4 subtiles x 16)
#define ROWS    144           // staged x rows (need 143 = 128 + 15)
#define XPITCH  144           // bytes per staged row (64*2B + 16B pad -> no LDS bank conflicts)

// workspace layout (bytes); total ~10.3 MB
#define WS_XHI   0
#define WS_XLO   (WS_XHI + B_SZ*S_LEN*D_IN*2)       //  4,096,000
#define WS_WPACK (WS_XLO + B_SZ*S_LEN*D_IN*2)       //  8,192,000
#define WS_BEFF  (WS_WPACK + 2*KDIM*H_OUT*2)        // 10,289,152

#define XS_BYTES   (2*ROWS*XPITCH)                  // 41,472
#define BT_UNITS   512                              // 16B units per B k-tile (32K x 64N x 2prec)
#define SMEM_BYTES (XS_BYTES + 2*BT_UNITS*16)       // 57,856  (< 64KB static limit)

#if defined(__AMDGCN__) && \
    __has_builtin(__builtin_amdgcn_global_load_async_to_lds_b128) && \
    __has_builtin(__builtin_amdgcn_s_wait_asynccnt)
#define USE_ASYNC_B 1
#else
#define USE_ASYNC_B 0
#endif

#if USE_ASYNC_B
// builtin signature (from hipcc diagnostic): (v4i AS1* src, v4i AS3* dst, imm, imm)
#define GPTR4(p) ((__attribute__((address_space(1))) v4i*)(p))
#define LPTR4(p) ((__attribute__((address_space(3))) v4i*)(p))
#endif

// ---------------------------------------------------------------------------
// Prologue 1: split x into bf16 hi/lo pair (hi = rne(x), lo = rne(x - hi))
// ---------------------------------------------------------------------------
__global__ void __launch_bounds__(256) prep_x(const float* __restrict__ x,
                                              __bf16* __restrict__ xhi,
                                              __bf16* __restrict__ xlo, int n) {
    int i = blockIdx.x * 256 + threadIdx.x;
    if (i < n) {
        float v  = x[i];
        __bf16 h = (__bf16)v;
        __bf16 l = (__bf16)(v - (float)h);
        xhi[i] = h;
        xlo[i] = l;
    }
}

// ---------------------------------------------------------------------------
// Prologue 2: build Weff, split hi/lo, store directly in the WMMA B-fragment
// lane layout (contiguous k order, same as the A reads): lane L = 16*half + n%16 holds
// elements i with K = kt*32 + 16*half + i, chunked as two 8-elem halves.
// Packed as 16B units: unit = (((prec*32+kt)*32+nt)*2+chunk)*32 + lane.
// ---------------------------------------------------------------------------
__global__ void __launch_bounds__(256) prep_w(const float* __restrict__ W0,
                                              const float* __restrict__ W1,
                                              const float* __restrict__ W2,
                                              __bf16* __restrict__ wpack) {
    int t = blockIdx.x * 256 + threadIdx.x;     // t < 512*1024
    int k = t & (KDIM - 1);
    int h = t >> 10;
    int j = k >> 6;
    float wv = W2[h * 1024 + k];
    if (j < 8) wv += W1[h * 512 + k];
    if (j < 4) wv += W0[h * 256 + k];
    wv *= (1.0f / 3.0f);
    __bf16 hi = (__bf16)wv;
    __bf16 lo = (__bf16)(wv - (float)hi);

    int kt = k >> 5, r = k & 31, half = r >> 4, i = r & 15;
    int chunk = i >> 3, e = i & 7;
    int nt = h >> 4, L = half * 16 + (h & 15);
    int base = (((kt * 32 + nt) * 2 + chunk) * 32 + L) * 8 + e;    // element index
    wpack[base]                = hi;                               // prec 0
    wpack[KDIM * H_OUT + base] = lo;                               // prec 1
}

// ---------------------------------------------------------------------------
// Prologue 3: fused bias
// ---------------------------------------------------------------------------
__global__ void __launch_bounds__(256) prep_b(const float* __restrict__ b0,
                                              const float* __restrict__ b1,
                                              const float* __restrict__ b2,
                                              float* __restrict__ beff) {
    int h = blockIdx.x * 256 + threadIdx.x;
    if (h < H_OUT) beff[h] = (b0[h] + b1[h] + b2[h]) * (1.0f / 3.0f);
}

// ---------------------------------------------------------------------------
// Main GEMM: one workgroup = 128(M) x 64(N) output block for one batch.
// 8 waves: wave w owns rows l0 + w*16 .. +15, all 64 N columns (4 subtiles).
// K loop: 32 steps of K=32, double-buffered B k-tiles in LDS, x slice in LDS.
// ---------------------------------------------------------------------------
__global__ void __launch_bounds__(256) patch_gemm(const __bf16* __restrict__ xhi,
                                                  const __bf16* __restrict__ xlo,
                                                  const __bf16* __restrict__ wpack,
                                                  const float*  __restrict__ beff,
                                                  const float*  __restrict__ pos,
                                                  float*        __restrict__ out) {
    __shared__ __align__(16) unsigned char smem[SMEM_BYTES];
    unsigned char* xs = smem;                                // [prec][ROWS][XPITCH]
    uint4* bt = (uint4*)(smem + XS_BYTES);                   // [2 buf][512] 16B units

    const int tid  = threadIdx.x;
    const int lane = tid & 31;
    const int w    = tid >> 5;          // wave 0..7
    const int m    = lane & 15;         // A row / C column within tile
    const int half = lane >> 4;
    const int nb = blockIdx.x;          // 0..7  -> N block
    const int mb = blockIdx.y;          // 0..7  -> M block
    const int b  = blockIdx.z;          // 0..31 -> batch
    const int l0  = mb * M_BLOCK;
    const int n0  = nb * N_BLOCK;
    const int nt0 = nb * 4;

    // ---- stage x slice (hi & lo) into LDS: 143 rows x 64 ch, padded pitch ----
    {
        const __bf16* srcs[2] = { xhi, xlo };
        for (int u = tid; u < 2 * ROWS * 8; u += 256) {
            int prec = u / (ROWS * 8);
            int rem  = u - prec * (ROWS * 8);
            int row  = rem >> 3;
            int ch   = rem & 7;
            int gr   = l0 + row;
            uint4 v = {0u, 0u, 0u, 0u};
            if (gr < S_LEN)
                v = *(const uint4*)(srcs[prec] + ((size_t)b * S_LEN + gr) * D_IN + ch * 8);
            *(uint4*)(xs + prec * (ROWS * XPITCH) + row * XPITCH + ch * 16) = v;
        }
    }

    // ---- B k-tile staging: 2 x 16B per thread, perfectly coalesced ----
    const uint4* wp4 = (const uint4*)wpack;
    auto gunit = [&](int kt, int u) -> int {
        int ln = u & 31, chunk = (u >> 5) & 1, ntl = (u >> 6) & 3, prec = u >> 8;
        return (((prec * 32 + kt) * 32 + (nt0 + ntl)) * 2 + chunk) * 32 + ln;
    };

#if USE_ASYNC_B
    // Direct global->LDS DMA (no VGPR staging, tracked by ASYNCcnt).
    auto issue_tile = [&](int kt, int buf) {
        __builtin_amdgcn_global_load_async_to_lds_b128(
            GPTR4(wp4 + gunit(kt, tid)),
            LPTR4(bt + buf * BT_UNITS + tid),
            0, 0);
        __builtin_amdgcn_global_load_async_to_lds_b128(
            GPTR4(wp4 + gunit(kt, tid + 256)),
            LPTR4(bt + buf * BT_UNITS + tid + 256),
            0, 0);
    };
    issue_tile(0, 0);
    __builtin_amdgcn_s_wait_asynccnt(0);
#else
    {
        uint4 r0 = wp4[gunit(0, tid)];
        uint4 r1 = wp4[gunit(0, tid + 256)];
        bt[tid]       = r0;
        bt[tid + 256] = r1;
    }
#endif
    __syncthreads();

    v8f acc[4] = {};

    for (int kt = 0; kt < 32; ++kt) {
        const int buf = kt & 1;

#if USE_ASYNC_B
        if (kt + 1 < 32) issue_tile(kt + 1, buf ^ 1);   // DMA overlaps WMMAs below
#else
        uint4 p0 = {0u,0u,0u,0u}, p1 = {0u,0u,0u,0u};
        if (kt + 1 < 32) {
            p0 = wp4[gunit(kt + 1, tid)];
            p1 = wp4[gunit(kt + 1, tid + 256)];
        }
#endif

        // ---- A fragments from LDS (contiguous k order, matches B pack) ----
        // element i<8 : K = kt*32 + 16*half + i        (one 16B ds_read)
        // element i>=8: K = kt*32 + 16*half + i        (second 16B, +16B)
        const int j    = kt >> 1;                      // tap index (fixed per kt)
        const int dA2  = ((kt & 1) * 32 + 16 * half) * 2;
        const int roff = (w * 16 + m + j) * XPITCH;
        const unsigned char* xrh = xs + roff;
        const unsigned char* xrl = xs + (ROWS * XPITCH) + roff;
        v8bf ah0 = *(const v8bf*)(xrh + dA2);
        v8bf ah1 = *(const v8bf*)(xrh + dA2 + 16);
        v8bf al0 = *(const v8bf*)(xrl + dA2);
        v8bf al1 = *(const v8bf*)(xrl + dA2 + 16);
        v16bf ahi = __builtin_shufflevector(ah0, ah1, 0,1,2,3,4,5,6,7,8,9,10,11,12,13,14,15);
        v16bf alo = __builtin_shufflevector(al0, al1, 0,1,2,3,4,5,6,7,8,9,10,11,12,13,14,15);

        // ---- 4 N-subtiles: 3 compensated WMMAs each, independent acc chains ----
        const uint4* btb = bt + buf * BT_UNITS;
#pragma unroll
        for (int ntl = 0; ntl < 4; ++ntl) {
            v8bf bh0 = *(const v8bf*)(btb + ((0 * 4 + ntl) * 2 + 0) * 32 + lane);
            v8bf bh1 = *(const v8bf*)(btb + ((0 * 4 + ntl) * 2 + 1) * 32 + lane);
            v8bf bl0 = *(const v8bf*)(btb + ((1 * 4 + ntl) * 2 + 0) * 32 + lane);
            v8bf bl1 = *(const v8bf*)(btb + ((1 * 4 + ntl) * 2 + 1) * 32 + lane);
            v16bf bhi = __builtin_shufflevector(bh0, bh1, 0,1,2,3,4,5,6,7,8,9,10,11,12,13,14,15);
            v16bf blo = __builtin_shufflevector(bl0, bl1, 0,1,2,3,4,5,6,7,8,9,10,11,12,13,14,15);
            acc[ntl] = __builtin_amdgcn_wmma_f32_16x16x32_bf16(
                false, alo, false, bhi, (short)0, acc[ntl], false, false);
            acc[ntl] = __builtin_amdgcn_wmma_f32_16x16x32_bf16(
                false, ahi, false, blo, (short)0, acc[ntl], false, false);
            acc[ntl] = __builtin_amdgcn_wmma_f32_16x16x32_bf16(
                false, ahi, false, bhi, (short)0, acc[ntl], false, false);
        }

#if USE_ASYNC_B
        __builtin_amdgcn_s_wait_asynccnt(0);   // next tile resident in LDS
#else
        if (kt + 1 < 32) {
            bt[(buf ^ 1) * BT_UNITS + tid]       = p0;
            bt[(buf ^ 1) * BT_UNITS + tid + 256] = p1;
        }
#endif
        __syncthreads();   // all waves done reading buf / next tile visible
    }

    // ---- epilogue: + beff + pos, masked store (C/D layout: N=lane%16, M=r+8*half) ----
#pragma unroll
    for (int ntl = 0; ntl < 4; ++ntl) {
        const int hcol = n0 + ntl * 16 + m;
        const float bb = beff[hcol];
#pragma unroll
        for (int r = 0; r < 8; ++r) {
            const int l = l0 + w * 16 + r + 8 * half;
            if (l < MINLEN) {
                out[((size_t)b * MINLEN + l) * H_OUT + hcol] =
                    acc[ntl][r] + bb + pos[(size_t)l * H_OUT + hcol];
            }
        }
    }
}

// ---------------------------------------------------------------------------
extern "C" void kernel_launch(void* const* d_in, const int* in_sizes, int n_in,
                              void* d_out, int out_size, void* d_ws, size_t ws_size,
                              hipStream_t stream) {
    (void)in_sizes; (void)n_in; (void)out_size; (void)ws_size;
    const float* x   = (const float*)d_in[0];
    const float* W0  = (const float*)d_in[1];
    const float* b0  = (const float*)d_in[2];
    const float* W1  = (const float*)d_in[3];
    const float* b1  = (const float*)d_in[4];
    const float* W2  = (const float*)d_in[5];
    const float* b2  = (const float*)d_in[6];
    const float* pos = (const float*)d_in[7];

    unsigned char* ws = (unsigned char*)d_ws;
    __bf16* xhi   = (__bf16*)(ws + WS_XHI);
    __bf16* xlo   = (__bf16*)(ws + WS_XLO);
    __bf16* wpack = (__bf16*)(ws + WS_WPACK);
    float*  beff  = (float*)(ws + WS_BEFF);
    float*  out   = (float*)d_out;

    const int nx = B_SZ * S_LEN * D_IN;                       // 2,048,000
    prep_x<<<(nx + 255) / 256, 256, 0, stream>>>(x, xhi, xlo, nx);
    prep_w<<<(H_OUT * KDIM) / 256, 256, 0, stream>>>(W0, W1, W2, wpack);
    prep_b<<<2, 256, 0, stream>>>(b0, b1, b2, beff);

    dim3 grid(H_OUT / N_BLOCK, (MINLEN + M_BLOCK - 1) / M_BLOCK, B_SZ);  // (8, 8, 32)
    patch_gemm<<<grid, 256, 0, stream>>>(xhi, xlo, wpack, beff, pos, out);
}
